// Decoder_481036337511
// MI455X (gfx1250) — matrix operand order repaired and checked by reading, not compiled, not run on hardware
//
#include <hip/hip_runtime.h>
#include <cstdint>

// ---------------- problem constants ----------------
#define T_T     1024            // T + PRE
#define N_POLES 4096
#define K_TOTAL 16385           // 1 + 4*N_POLES
#define K_PAD   16416           // 513 * 32
#define N_KT    513             // K_PAD / 32
#define D_DIM   2048
#define B_SZ    4
#define LDSROW  132             // 128 + pad for fp32 tile rows (16B-aligned rows)
#define TILE_F  (32 * LDSROW)   // floats per fp32 LDS x-tile buffer
#define BCOL    40              // bf16 tile column stride (80 B: 16B-aligned, bank-friendly)
#define A_ELEMS ((size_t)64 * N_KT * 512)   // 1024*16416 elements, fragment-swizzled

typedef __attribute__((ext_vector_type(16))) __bf16 v16bf;
typedef __attribute__((ext_vector_type(8)))  float  v8f;
typedef int v4i __attribute__((vector_size(16)));   // matches async builtin's pointee type

#if defined(__AMDGCN__) && __has_builtin(__builtin_amdgcn_global_load_async_to_lds_b128)
#define USE_ASYNC 1
#else
#define USE_ASYNC 0
#endif

// ================= pass 1: column L2 norms =================
__global__ __launch_bounds__(256) void dict_norm_kernel(const float* __restrict__ rr,
                                                        const float* __restrict__ theta,
                                                        float* __restrict__ G)
{
    int k = blockIdx.x * 256 + threadIdx.x;
    if (k >= K_PAD) return;
    if (k >= K_TOTAL) { G[k] = 1.0f; return; }       // padding columns (values are 0)
    if (k == 0)       { G[0] = 32.0f; return; }      // sqrt(1024)

    int g = (k - 1) >> 12;            // 0: pw*c, 1: sign*pw*c, 2: pw*s, 3: sign*pw*s
    int j = (k - 1) & (N_POLES - 1);
    float r = rr[j], th = theta[j];
    float pw = 1.0f, sumsq = 0.0f;
    for (int t = 0; t < T_T; ++t) {
        float s, c;
        sincosf((float)t * th, &s, &c);
        float base = pw * ((g < 2) ? c : s);          // sign^2 == 1, irrelevant for norm
        sumsq = fmaf(base, base, sumsq);
        pw *= r;
    }
    float n = sqrtf(sumsq);
    G[k] = (n == 0.0f) ? 32.0f : n;
}

// ===== pass 2: build normalized dictionary in WMMA-A fragment layout, bf16 hi/lo split =====
// A fragment layout (16-bit 16x32 A-matrix, wave32, contiguous K as in B):
//   lane<16 : row M = lane,    slots 0..15 -> K 0..15
//   lane>=16: row M = lane-16, slots 0..15 -> K 16..31
__global__ __launch_bounds__(256) void dict_build_kernel(const float* __restrict__ rr,
                                                         const float* __restrict__ theta,
                                                         const float* __restrict__ G,
                                                         __bf16* __restrict__ Ahi,
                                                         __bf16* __restrict__ Alo)
{
    size_t idx = (size_t)blockIdx.x * 256 + threadIdx.x;   // < A_ELEMS
    int frag   = (int)(idx >> 9);
    int within = (int)(idx & 511);
    int lane = within >> 4;
    int slot = within & 15;
    int mt = frag / N_KT;
    int kt = frag - mt * N_KT;
    int half = lane >> 4;
    int m  = lane & 15;
    int kk = (half << 4) | slot;
    int t = mt * 16 + m;
    int k = kt * 32 + kk;

    float val = 0.0f;
    if (k < K_TOTAL) {
        if (k == 0) {
            val = 1.0f;
        } else {
            int g = (k - 1) >> 12;
            int j = (k - 1) & (N_POLES - 1);
            float r = rr[j], th = theta[j];
            float pw = powf(r, (float)t);
            float s, c;
            sincosf((float)t * th, &s, &c);
            float base = pw * ((g < 2) ? c : s);
            if ((g & 1) && (t & 1)) base = -base;     // sign = (-1)^t for groups 1,3
            val = base;
        }
        val /= G[k];
    }
    __bf16 hi = (__bf16)val;
    __bf16 lo = (__bf16)(val - (float)hi);
    Ahi[idx] = hi;
    Alo[idx] = lo;
}

// ================= pass 3: GEMM out[b,t,d] = sum_k dic[t,k]*x[b,k,d] =================
__device__ __forceinline__ void issue_tile(float* __restrict__ lbuf,
                                           const float* __restrict__ xbase,
                                           int k0, int d0, int tid)
{
    // 256 threads cooperatively fetch a 32(k) x 128(d) fp32 tile; 16 floats/thread.
    int row = tid >> 3;
    int col = (tid & 7) * 16;
    int gk = k0 + row;
    if (gk > K_TOTAL - 1) gk = K_TOTAL - 1;           // clamp (A column is 0 there)
    const float* gp = xbase + (size_t)gk * D_DIM + d0 + col;
    float* lp = lbuf + row * LDSROW + col;
#if USE_ASYNC
    #pragma unroll
    for (int i = 0; i < 4; ++i) {
        __builtin_amdgcn_global_load_async_to_lds_b128(
            (__attribute__((address_space(1))) v4i*)(gp + i * 4),
            (__attribute__((address_space(3))) v4i*)(lp + i * 4),
            0, 0);
    }
#else
    #pragma unroll
    for (int i = 0; i < 4; ++i)
        *(float4*)(lp + i * 4) = *(const float4*)(gp + i * 4);
#endif
}

__device__ __forceinline__ void wait_async_le4() {
#if USE_ASYNC
    asm volatile("s_wait_asynccnt 0x4" ::: "memory");
#endif
}
__device__ __forceinline__ void wait_async_le0() {
#if USE_ASYNC
    asm volatile("s_wait_asynccnt 0x0" ::: "memory");
#endif
}

__global__ __launch_bounds__(256) void dict_gemm_kernel(const float* __restrict__ x,
                                                        const __bf16* __restrict__ Ahi,
                                                        const __bf16* __restrict__ Alo,
                                                        float* __restrict__ out)
{
    __shared__ float  fbuf[2 * TILE_F];               // double-buffered fp32 x tile (~33 KB)
    __shared__ __bf16 hbuf[128 * BCOL];               // bf16 hi tile, [col][k] layout (10 KB)
    __shared__ __bf16 lbuf[128 * BCOL];               // bf16 lo tile, [col][k] layout (10 KB)

    const int tid  = threadIdx.x;
    const int wave = tid >> 5;
    const int lane = tid & 31;
    const int wm = wave & 3;                          // 4 M-groups of 32 rows
    const int wn = wave >> 2;                         // 2 N-groups of 64 cols
    const int Mblk = blockIdx.y * 128;                // grid.y = 8
    const int Cblk = blockIdx.x * 128;                // grid.x = 64 (over B*D = 8192)
    const int bb = Cblk >> 11;                        // batch
    const int d0 = Cblk & (D_DIM - 1);
    const float* xbase = x + (size_t)bb * K_TOTAL * D_DIM;

    v8f acc[2][4] = {};

    issue_tile(fbuf, xbase, 0, d0, tid);

    const int n    = lane & 15;                       // B-frag column within 16
    const int kset = (lane >> 4) * 16;                // B-frag K half (0 or 16)
    const int ccol  = tid & 127;                      // convert-phase column
    const int khalf = (tid >> 7) * 16;                // convert-phase K half

    for (int kt = 0; kt < N_KT; ++kt) {
        const int cur = kt & 1;
        if (kt + 1 < N_KT) {
            issue_tile(fbuf + (1 - cur) * TILE_F, xbase, (kt + 1) * 32, d0, tid);
            wait_async_le4();                         // previous tile's 4 loads done
        } else {
            wait_async_le0();
        }
        __syncthreads();                              // tile-kt fp32 data in LDS (all waves)

        // A fragments: contiguous, pre-swizzled, L2-resident (issued early to overlap)
        v16bf ahi[2], alo[2];
        #pragma unroll
        for (int i = 0; i < 2; ++i) {
            const int mt = (Mblk >> 4) + wm * 2 + i;
            const size_t off = ((size_t)(mt * N_KT + kt)) * 512 + (size_t)lane * 16;
            ahi[i] = *(const v16bf*)(Ahi + off);
            alo[i] = *(const v16bf*)(Alo + off);
        }

        // Shared convert phase: fp32 tile -> bf16 hi/lo tiles, transposed to [col][k]
        // so each B fragment is a single contiguous 32B load per lane.
        {
            const float* fp = fbuf + cur * TILE_F + ccol;
            v16bf hv, lv;
            #pragma unroll
            for (int e = 0; e < 16; ++e) {
                float v = fp[(khalf + e) * LDSROW];
                __bf16 h = (__bf16)v;
                hv[e] = h;
                lv[e] = (__bf16)(v - (float)h);
            }
            *(v16bf*)(&hbuf[ccol * BCOL + khalf]) = hv;
            *(v16bf*)(&lbuf[ccol * BCOL + khalf]) = lv;
        }
        __syncthreads();                              // converted tiles visible

        #pragma unroll
        for (int j = 0; j < 4; ++j) {
            const int col = wn * 64 + j * 16 + n;
            v16bf bhi = *(const v16bf*)(&hbuf[col * BCOL + kset]);
            v16bf blo = *(const v16bf*)(&lbuf[col * BCOL + kset]);
            #pragma unroll
            for (int i = 0; i < 2; ++i) {
                // bf16x3: hi*hi + hi*lo + lo*hi, fp32 accumulate
                acc[i][j] = __builtin_amdgcn_wmma_f32_16x16x32_bf16(
                    false, ahi[i], false, bhi, (short)0, acc[i][j], false, false);
                acc[i][j] = __builtin_amdgcn_wmma_f32_16x16x32_bf16(
                    false, ahi[i], false, blo, (short)0, acc[i][j], false, false);
                acc[i][j] = __builtin_amdgcn_wmma_f32_16x16x32_bf16(
                    false, alo[i], false, bhi, (short)0, acc[i][j], false, false);
            }
        }
        __syncthreads();                              // compute done before next convert/refill
    }

    // Epilogue: C/D layout -> lanes 0-15: M=vgpr, lanes 16-31: M=vgpr+8, N=lane&15
    #pragma unroll
    for (int i = 0; i < 2; ++i) {
        const int tbase = Mblk + wm * 32 + i * 16 + (lane >> 4) * 8;
        #pragma unroll
        for (int j = 0; j < 4; ++j) {
            const int dcol = d0 + wn * 64 + j * 16 + n;
            #pragma unroll
            for (int vg = 0; vg < 8; ++vg) {
                out[((size_t)bb * T_T + tbase + vg) * D_DIM + dcol] = acc[i][j][vg];
            }
        }
    }
}

// ================= launcher =================
extern "C" void kernel_launch(void* const* d_in, const int* in_sizes, int n_in,
                              void* d_out, int out_size, void* d_ws, size_t ws_size,
                              hipStream_t stream)
{
    (void)in_sizes; (void)n_in; (void)out_size; (void)ws_size;

    const float* rr    = (const float*)d_in[0];
    const float* theta = (const float*)d_in[1];
    const float* x     = (const float*)d_in[2];
    float* out = (float*)d_out;

    // workspace: G (K_PAD f32) | A_hi (A_ELEMS bf16) | A_lo (A_ELEMS bf16)  ~67.3 MB
    float*  G   = (float*)d_ws;
    __bf16* Ahi = (__bf16*)((char*)d_ws + (size_t)K_PAD * sizeof(float));
    __bf16* Alo = Ahi + A_ELEMS;

    dict_norm_kernel<<<(K_PAD + 255) / 256, 256, 0, stream>>>(rr, theta, G);
    dict_build_kernel<<<(unsigned)(A_ELEMS / 256), 256, 0, stream>>>(rr, theta, G, Ahi, Alo);
    dim3 grid(64, 8);
    dict_gemm_kernel<<<grid, 256, 0, stream>>>(x, Ahi, Alo, out);
}
